// DeformableTransformerDecoderLayer_3126736191886
// MI455X (gfx1250) — matrix operand order repaired and checked by reading, not compiled, not run on hardware
//
#include <hip/hip_runtime.h>
#include <hip/hip_bf16.h>

// ---------------------------------------------------------------------------
// Deformable Transformer Decoder Layer — CDNA5 (gfx1250) WMMA implementation
//
//  1. qkv    = tgt @ in_proj^T + b                      (WMMA GEMM, async-LDS)
//  2. sa     = flash-attention(qkv)                     (fused WMMA kernel)
//  3. tmp    = sa @ out_proj^T + b + tgt                (WMMA GEMM, resid)
//  4. tgt1   = LN1(tmp)
//  5. value  = mask0(memory @ value_proj^T + b) -> f16  (WMMA GEMM, f16 out)
//  6. off    = tgt1 @ samp_off^T + b                    (WMMA GEMM)
//  7. awl    = tgt1 @ attn_w^T + b                      (WMMA GEMM)
//  8. ca     = msdeform_sample(value, off, softmax(awl), ref)   (gather)
//  9. tmp    = ca @ output_proj^T + b + tgt1            (WMMA GEMM, resid)
// 10. tgt2   = LN2(tmp)
// 11. hid    = relu(tgt2 @ lin1^T + b)                  (WMMA GEMM, relu)
// 12. tmp    = hid @ lin2^T + b + tgt2                  (WMMA GEMM, resid)
// 13. out    = LN3(tmp)
// ---------------------------------------------------------------------------

typedef __attribute__((ext_vector_type(16))) _Float16 v16h;
typedef __attribute__((ext_vector_type(8)))  float    v8f;
typedef int v4i_vs __attribute__((vector_size(16)));   // matches builtin param

#define NB      8            // batch
#define NQ      900          // queries
#define DMODEL  256
#define NHEADS  8
#define HDIM    32
#define DFFN    1024
#define STOT    13294        // 100*100 + 50*50 + 25*25 + 13*13
#define BQ      (NB * NQ)    // 7200
#define KSTR    34           // padded f16 K-stride in LDS (attention kernel)
#define KSTR32  36           // padded f32 K-stride (GEMM; rows stay 16B aligned)

#if __has_builtin(__builtin_amdgcn_global_load_async_to_lds_b128) && \
    __has_builtin(__builtin_amdgcn_s_wait_asynccnt)
#define HAVE_GFX1250_ASYNC 1
#else
#define HAVE_GFX1250_ASYNC 0
#endif

__device__ __forceinline__ void async_cp16(const float* g, float* l) {
#if HAVE_GFX1250_ASYNC
  __builtin_amdgcn_global_load_async_to_lds_b128(
      (__attribute__((address_space(1))) v4i_vs*)(v4i_vs*)(void*)g,
      (__attribute__((address_space(3))) v4i_vs*)(v4i_vs*)(void*)l, 0, 0);
#else
  (void)g; (void)l;
#endif
}

__device__ __forceinline__ v8f wmma_f16(v16h a, v16h b, v8f c) {
  return __builtin_amdgcn_wmma_f32_16x16x32_f16(false, a, false, b, (short)0, c,
                                                false, false);
}

// A fragment: 16 rows (M) x 32 (K), from row-major f16 LDS region.
// ISA 7.12.2: lanes 0-15 M=0..15  VGPR0-3: K=0..7, VGPR4-7: K=16..23
//             lanes 16-31 M=0..15 VGPR0-3: K=8..15, VGPR4-7: K=24..31
__device__ __forceinline__ v16h frag_a(const _Float16* base, int stride, int lane) {
  const int half = (lane >> 4) & 1;
  const _Float16* row = base + (lane & 15) * stride;
  v16h f;
#pragma unroll
  for (int v = 0; v < 8; ++v) {
    const int k = ((v < 4) ? 0 : 16) + half * 8 + (v & 3) * 2;
    f[2 * v]     = row[k];
    f[2 * v + 1] = row[k + 1];
  }
  return f;
}

// B fragment: 32 (K) x 16 cols (N); source region stored [n][k] row-major.
// Same K order as A: lanes 0-15 VGPR0-3: K=0..7, VGPR4-7: K=16..23; lanes 16-31 K=8..15, 24..31.
__device__ __forceinline__ v16h frag_b(const _Float16* base, int stride, int lane) {
  const int half = (lane >> 4) & 1;
  const _Float16* row = base + (lane & 15) * stride;
  v16h f;
#pragma unroll
  for (int v = 0; v < 8; ++v) {
    const int k = ((v < 4) ? 0 : 16) + half * 8 + (v & 3) * 2;
    f[2 * v]     = row[k];
    f[2 * v + 1] = row[k + 1];
  }
  return f;
}

// Same fragment loaders but reading raw fp32 LDS tiles (async-copied bytes),
// converting to f16 at load time.
__device__ __forceinline__ v16h frag_a32(const float* base, int stride, int lane) {
  const int half = (lane >> 4) & 1;
  const float* row = base + (lane & 15) * stride;
  v16h f;
#pragma unroll
  for (int v = 0; v < 8; ++v) {
    const int k = ((v < 4) ? 0 : 16) + half * 8 + (v & 3) * 2;
    f[2 * v]     = (_Float16)row[k];
    f[2 * v + 1] = (_Float16)row[k + 1];
  }
  return f;
}

__device__ __forceinline__ v16h frag_b32(const float* base, int stride, int lane) {
  const int half = (lane >> 4) & 1;
  const float* row = base + (lane & 15) * stride;
  v16h f;
#pragma unroll
  for (int v = 0; v < 8; ++v) {
    const int k = ((v < 4) ? 0 : 16) + half * 8 + (v & 3) * 2;
    f[2 * v]     = (_Float16)row[k];
    f[2 * v + 1] = (_Float16)row[k + 1];
  }
  return f;
}

// ---------------------------------------------------------------------------
// Generic WMMA GEMM:  out[M,N] = epilogue( A[M,K] @ W[N,K]^T + bias[N] )
// 64x64 tile per block (128 threads = 4 waves, each wave: 16 rows x 64 cols).
// Double-buffered async global->LDS staging (ASYNCcnt pipelined); OOB M-rows
// are source-clamped so each wave issues exactly 8 async b128 per tile and
// the s_wait_asynccnt threshold is exact.  N % 64 == 0, K % 32 == 0.
// ---------------------------------------------------------------------------
template <bool RELU, bool RESID, bool F16OUT, bool MASK>
__global__ __launch_bounds__(128) void gemm_wmma_kernel(
    const float* __restrict__ A, const float* __restrict__ W,
    const float* __restrict__ bias, const float* __restrict__ resid,
    const unsigned char* __restrict__ mask, float* __restrict__ outF,
    _Float16* __restrict__ outH, int M, int N, int K) {
  __shared__ float As[2][64 * KSTR32];
  __shared__ float Bs[2][64 * KSTR32];

  const int t    = threadIdx.x;
  const int lane = t & 31;
  const int wave = t >> 5;
  const int half = (lane >> 4) & 1;
  const int n    = lane & 15;
  const int row0 = blockIdx.y * 64;
  const int col0 = blockIdx.x * 64;
  const int ntiles = K >> 5;

  v8f acc[4] = {};

  auto stage = [&](int ti, int buf) {
#pragma unroll
    for (int i = 0; i < 4; ++i) {
      const int j = t + i * 128;        // [0,512) float4 chunks of the 64x32 tile
      const int r = j >> 3;             // tile row
      const int c = (j & 7) * 4;        // tile k (16B aligned)
      int gr = row0 + r;
      if (gr >= M) gr = M - 1;          // clamp: bad rows' outputs are discarded
      const float* ga = A + (size_t)gr * K + ti * 32 + c;
      const float* gb = W + (size_t)(col0 + r) * K + ti * 32 + c;
      float* la = &As[buf][r * KSTR32 + c];
      float* lb = &Bs[buf][r * KSTR32 + c];
#if HAVE_GFX1250_ASYNC
      async_cp16(ga, la);
      async_cp16(gb, lb);
#else
      const float4 av = *reinterpret_cast<const float4*>(ga);
      const float4 bv = *reinterpret_cast<const float4*>(gb);
      la[0] = av.x; la[1] = av.y; la[2] = av.z; la[3] = av.w;
      lb[0] = bv.x; lb[1] = bv.y; lb[2] = bv.z; lb[3] = bv.w;
#endif
    }
  };

  stage(0, 0);
  for (int ti = 0; ti < ntiles; ++ti) {
    const int buf = ti & 1;
    if (ti + 1 < ntiles) {
      stage(ti + 1, buf ^ 1);           // prefetch next tile into other buffer
#if HAVE_GFX1250_ASYNC
      __builtin_amdgcn_s_wait_asynccnt(8);   // current tile done, next in flight
#endif
    } else {
#if HAVE_GFX1250_ASYNC
      __builtin_amdgcn_s_wait_asynccnt(0);
#endif
    }
    __syncthreads();

    const v16h af = frag_a32(&As[buf][wave * 16 * KSTR32], KSTR32, lane);
#pragma unroll
    for (int nt = 0; nt < 4; ++nt) {
      const v16h bf = frag_b32(&Bs[buf][nt * 16 * KSTR32], KSTR32, lane);
      acc[nt] = wmma_f16(af, bf, acc[nt]);
    }
    __syncthreads();
  }

  // Epilogue.  C layout (ISA 7.12.2): elem r of v8f @ lane -> M = r + half*8,
  // N = lane&15.
#pragma unroll
  for (int nt = 0; nt < 4; ++nt) {
    const int gc = col0 + nt * 16 + n;
    const float bv = bias[gc];
#pragma unroll
    for (int r = 0; r < 8; ++r) {
      const int gr = row0 + wave * 16 + half * 8 + r;
      if (gr < M) {
        float v = acc[nt][r] + bv;
        if (MASK) { if (mask[gr]) v = 0.f; }
        if (RESID) v += resid[(size_t)gr * N + gc];
        if (RELU)  v = fmaxf(v, 0.f);
        if (F16OUT) outH[(size_t)gr * N + gc] = (_Float16)v;
        else        outF[(size_t)gr * N + gc] = v;
      }
    }
  }
}

// ---------------------------------------------------------------------------
// Fused flash self-attention.  One block = 64 queries of one (b,h); 4 waves,
// each wave owns 16 queries.  HEAD_DIM = 32 == WMMA K, so one WMMA per
// 16x16 score tile and two per P*V step.  Online softmax, no score matrix.
// ---------------------------------------------------------------------------
__global__ __launch_bounds__(128) void attn_flash_kernel(
    const float* __restrict__ qkv, float* __restrict__ sa) {
  __shared__ _Float16 Ks[32 * KSTR];       // [key][d]
  __shared__ _Float16 Vs[32 * KSTR];       // [d][key]   (transposed)
  __shared__ _Float16 Ps[4 * 16 * KSTR];   // per-wave P tile [m][k]

  const int t    = threadIdx.x;
  const int lane = t & 31;
  const int wave = t >> 5;
  const int half = (lane >> 4) & 1;
  const int n    = lane & 15;
  const int b    = blockIdx.y >> 3;
  const int h    = blockIdx.y & 7;
  const int q0   = blockIdx.x * 64 + wave * 16;
  const float scale = 0.17677669529663687f;  // 1/sqrt(32)

  // Q fragment loaded straight from global (scaled, f32->f16).
  v16h qf;
  {
    const int m  = lane & 15;
    const int q  = q0 + m;
    const bool ok = q < NQ;
    const float* qr = qkv + (size_t)(b * NQ + (ok ? q : 0)) * 768 + h * HDIM;
#pragma unroll
    for (int v = 0; v < 8; ++v) {
      const int k = ((v < 4) ? 0 : 16) + half * 8 + (v & 3) * 2;
      qf[2 * v]     = (_Float16)(ok ? qr[k] * scale : 0.f);
      qf[2 * v + 1] = (_Float16)(ok ? qr[k + 1] * scale : 0.f);
    }
  }

  float run_max[8], run_sum[8];
#pragma unroll
  for (int r = 0; r < 8; ++r) { run_max[r] = -1e30f; run_sum[r] = 0.f; }
  v8f o0 = {}, o1 = {};

  for (int kt = 0; kt < NQ; kt += 32) {
    // Stage 32 keys of K and V (f16), V transposed for B-fragment loads.
#pragma unroll
    for (int i = 0; i < 8; ++i) {
      const int e   = t + i * 128;   // [0,1024)
      const int key = e >> 5;
      const int d   = e & 31;
      const int gk  = kt + key;
      float kv = 0.f, vv = 0.f;
      if (gk < NQ) {
        const float* rp = qkv + (size_t)(b * NQ + gk) * 768 + h * HDIM;
        kv = rp[256 + d];
        vv = rp[512 + d];
      }
      Ks[key * KSTR + d] = (_Float16)kv;
      Vs[d * KSTR + key] = (_Float16)vv;
    }
    __syncthreads();

    const v16h kf0 = frag_b(Ks, KSTR, lane);
    const v16h kf1 = frag_b(Ks + 16 * KSTR, KSTR, lane);
    v8f s0 = {}, s1 = {};
    s0 = wmma_f16(qf, kf0, s0);
    s1 = wmma_f16(qf, kf1, s1);

    // mask out-of-range key columns
    if (kt + n >= NQ) {
#pragma unroll
      for (int r = 0; r < 8; ++r) s0[r] = -1e30f;
    }
    if (kt + 16 + n >= NQ) {
#pragma unroll
      for (int r = 0; r < 8; ++r) s1[r] = -1e30f;
    }

    float corr[8];
#pragma unroll
    for (int r = 0; r < 8; ++r) {
      float mx = fmaxf(s0[r], s1[r]);
      for (int o = 1; o < 16; o <<= 1) mx = fmaxf(mx, __shfl_xor(mx, o, 32));
      const float nm = fmaxf(run_max[r], mx);
      corr[r] = __expf(run_max[r] - nm);
      run_max[r] = nm;
      const float p0 = __expf(s0[r] - nm);
      const float p1 = __expf(s1[r] - nm);
      s0[r] = p0; s1[r] = p1;
      float rs = p0 + p1;
      for (int o = 1; o < 16; o <<= 1) rs += __shfl_xor(rs, o, 32);
      run_sum[r] = run_sum[r] * corr[r] + rs;
    }

    // C-layout -> A-layout round trip through LDS for the P fragment.
    _Float16* P = Ps + wave * 16 * KSTR;
#pragma unroll
    for (int r = 0; r < 8; ++r) {
      const int m = half * 8 + r;
      P[m * KSTR + n]      = (_Float16)s0[r];
      P[m * KSTR + 16 + n] = (_Float16)s1[r];
    }
    const v16h pa  = frag_a(P, KSTR, lane);
    const v16h vf0 = frag_b(Vs, KSTR, lane);
    const v16h vf1 = frag_b(Vs + 16 * KSTR, KSTR, lane);
#pragma unroll
    for (int r = 0; r < 8; ++r) { o0[r] *= corr[r]; o1[r] *= corr[r]; }
    o0 = wmma_f16(pa, vf0, o0);
    o1 = wmma_f16(pa, vf1, o1);
    __syncthreads();
  }

#pragma unroll
  for (int r = 0; r < 8; ++r) {
    const int q = q0 + half * 8 + r;
    if (q < NQ) {
      const float inv = 1.f / run_sum[r];
      float* orow = sa + (size_t)(b * NQ + q) * DMODEL + h * HDIM;
      orow[n]      = o0[r] * inv;
      orow[16 + n] = o1[r] * inv;
    }
  }
}

// ---------------------------------------------------------------------------
// LayerNorm over D=256: one wave per row, 8 rows per 256-thread block.
// ---------------------------------------------------------------------------
__global__ __launch_bounds__(256) void layernorm_kernel(
    const float* __restrict__ x, const float* __restrict__ g,
    const float* __restrict__ bta, float* __restrict__ y, int M) {
  const int wave = threadIdx.x >> 5;
  const int lane = threadIdx.x & 31;
  const int row  = blockIdx.x * 8 + wave;
  if (row >= M) return;
  const float* xr = x + (size_t)row * DMODEL;
  float v[8];
  float s = 0.f;
#pragma unroll
  for (int i = 0; i < 8; ++i) { v[i] = xr[lane + i * 32]; s += v[i]; }
  for (int o = 1; o < 32; o <<= 1) s += __shfl_xor(s, o, 32);
  const float mu = s * (1.f / 256.f);
  float qs = 0.f;
#pragma unroll
  for (int i = 0; i < 8; ++i) { const float d = v[i] - mu; qs += d * d; }
  for (int o = 1; o < 32; o <<= 1) qs += __shfl_xor(qs, o, 32);
  const float inv = rsqrtf(qs * (1.f / 256.f) + 1e-5f);
  float* yr = y + (size_t)row * DMODEL;
#pragma unroll
  for (int i = 0; i < 8; ++i) {
    const int c = lane + i * 32;
    yr[c] = (v[i] - mu) * inv * g[c] + bta[c];
  }
}

// ---------------------------------------------------------------------------
// Multi-scale deformable sampling.  One block per (b,q); thread = (head,chan):
// lanes of a wave span the 32 channels of one head -> each bilinear tap is a
// coalesced 64 B f16 read.  Scalar point math is replicated per lane (cheap).
// ---------------------------------------------------------------------------
__global__ __launch_bounds__(256) void deform_kernel(
    const float* __restrict__ ref, const _Float16* __restrict__ value,
    const float* __restrict__ off, const float* __restrict__ awl,
    float* __restrict__ out) {
  const int bq = blockIdx.x;
  const int b  = bq / NQ;
  const int h  = threadIdx.x >> 5;
  const int c  = threadIdx.x & 31;

  const int Hs[4] = {100, 50, 25, 13};
  const int Ws[4] = {100, 50, 25, 13};
  const int S0[4] = {0, 10000, 12500, 13125};

  // softmax over the 16 (level,point) attention logits of this head
  const float* awr = awl + (size_t)bq * 128 + h * 16;
  float mx = -1e30f;
#pragma unroll
  for (int j = 0; j < 16; ++j) mx = fmaxf(mx, awr[j]);
  float aw[16];
  float den = 0.f;
#pragma unroll
  for (int j = 0; j < 16; ++j) { aw[j] = __expf(awr[j] - mx); den += aw[j]; }
  const float dinv = 1.f / den;

  const float* offr = off + (size_t)bq * 256 + h * 32;
  float outacc = 0.f;
#pragma unroll
  for (int l = 0; l < 4; ++l) {
    const int   Wl = Ws[l], Hl = Hs[l];
    const float rx = ref[((size_t)bq * 4 + l) * 2 + 0];
    const float ry = ref[((size_t)bq * 4 + l) * 2 + 1];
    const _Float16* vbase =
        value + ((size_t)(b * STOT + S0[l])) * DMODEL + h * HDIM + c;
#pragma unroll
    for (int p = 0; p < 4; ++p) {
      const float ox = offr[l * 8 + p * 2 + 0];
      const float oy = offr[l * 8 + p * 2 + 1];
      const float x = (rx + ox / (float)Wl) * (float)Wl - 0.5f;
      const float y = (ry + oy / (float)Hl) * (float)Hl - 0.5f;
      const float x0f = floorf(x), y0f = floorf(y);
      float acc = 0.f;
#pragma unroll
      for (int dy = 0; dy < 2; ++dy) {
#pragma unroll
        for (int dx = 0; dx < 2; ++dx) {
          const float xi = x0f + (float)dx;
          const float yi = y0f + (float)dy;
          if (xi >= 0.f && xi < (float)Wl && yi >= 0.f && yi < (float)Hl) {
            const float w = (1.f - fabsf(x - xi)) * (1.f - fabsf(y - yi));
            const int idx = (int)yi * Wl + (int)xi;
            acc += w * (float)vbase[(size_t)idx * DMODEL];
          }
        }
      }
      outacc += aw[l * 4 + p] * dinv * acc;
    }
  }
  out[(size_t)bq * DMODEL + h * HDIM + c] = outacc;
}

// ---------------------------------------------------------------------------
// Host side
// ---------------------------------------------------------------------------
template <bool RELU, bool RESID, bool F16OUT, bool MASK>
static void launch_gemm(const float* A, const float* W, const float* bias,
                        const float* resid, const unsigned char* mask,
                        void* out, int M, int N, int K, hipStream_t s) {
  dim3 grid(N / 64, (M + 63) / 64);
  gemm_wmma_kernel<RELU, RESID, F16OUT, MASK><<<grid, 128, 0, s>>>(
      A, W, bias, resid, mask, (float*)out, (_Float16*)out, M, N, K);
}

extern "C" void kernel_launch(void* const* d_in, const int* in_sizes, int n_in,
                              void* d_out, int out_size, void* d_ws,
                              size_t ws_size, hipStream_t stream) {
  const float* tgt       = (const float*)d_in[0];
  const float* memory    = (const float*)d_in[1];
  const unsigned char* pmask = (const unsigned char*)d_in[2];
  const float* refp      = (const float*)d_in[3];
  // d_in[4], d_in[5]: spatial shapes / level starts (static, hardcoded)
  const float* in_proj_w = (const float*)d_in[6];
  const float* in_proj_b = (const float*)d_in[7];
  const float* out_proj_w= (const float*)d_in[8];
  const float* out_proj_b= (const float*)d_in[9];
  const float* n1g = (const float*)d_in[10]; const float* n1b = (const float*)d_in[11];
  const float* n2g = (const float*)d_in[12]; const float* n2b = (const float*)d_in[13];
  const float* n3g = (const float*)d_in[14]; const float* n3b = (const float*)d_in[15];
  const float* so_w = (const float*)d_in[16]; const float* so_b = (const float*)d_in[17];
  const float* aw_w = (const float*)d_in[18]; const float* aw_b = (const float*)d_in[19];
  const float* vp_w = (const float*)d_in[20]; const float* vp_b = (const float*)d_in[21];
  const float* op_w = (const float*)d_in[22]; const float* op_b = (const float*)d_in[23];
  const float* l1_w = (const float*)d_in[24]; const float* l1_b = (const float*)d_in[25];
  const float* l2_w = (const float*)d_in[26]; const float* l2_b = (const float*)d_in[27];

  char* ws = (char*)d_ws;
  const size_t o_qkv  = 0;                               // 7200*768 f32
  const size_t o_sa   = o_qkv  + (size_t)BQ * 768 * 4;   // 7200*256 f32
  const size_t o_tmp  = o_sa   + (size_t)BQ * DMODEL * 4;
  const size_t o_tgt1 = o_tmp  + (size_t)BQ * DMODEL * 4;
  const size_t o_tgt2 = o_tgt1 + (size_t)BQ * DMODEL * 4;
  const size_t o_val  = o_tgt2 + (size_t)BQ * DMODEL * 4;      // f16 value
  const size_t o_off  = o_val  + (size_t)NB * STOT * DMODEL * 2;
  const size_t o_awl  = o_off  + (size_t)BQ * 256 * 4;
  const size_t o_ca   = o_awl  + (size_t)BQ * 128 * 4;
  const size_t o_hid  = o_ca   + (size_t)BQ * DMODEL * 4;      // 7200*1024 f32

  float*    qkv  = (float*)(ws + o_qkv);
  float*    sa   = (float*)(ws + o_sa);
  float*    tmp  = (float*)(ws + o_tmp);
  float*    tgt1 = (float*)(ws + o_tgt1);
  float*    tgt2 = (float*)(ws + o_tgt2);
  _Float16* valh = (_Float16*)(ws + o_val);
  float*    offb = (float*)(ws + o_off);
  float*    awlb = (float*)(ws + o_awl);
  float*    ca   = (float*)(ws + o_ca);
  float*    hid  = (float*)(ws + o_hid);
  float*    outp = (float*)d_out;

  const int MV = NB * STOT;  // 106352
  const dim3 lnGrid((BQ + 7) / 8);

  // 1. qkv projection
  launch_gemm<false, false, false, false>(tgt, in_proj_w, in_proj_b, nullptr,
                                          nullptr, qkv, BQ, 768, DMODEL, stream);
  // 2. fused flash self-attention
  attn_flash_kernel<<<dim3((NQ + 63) / 64, NB * NHEADS), 128, 0, stream>>>(qkv, sa);
  // 3. out proj + residual(tgt)
  launch_gemm<false, true, false, false>(sa, out_proj_w, out_proj_b, tgt,
                                         nullptr, tmp, BQ, DMODEL, DMODEL, stream);
  // 4. LN1 -> tgt1
  layernorm_kernel<<<lnGrid, 256, 0, stream>>>(tmp, n1g, n1b, tgt1, BQ);
  // 5. value projection (masked, f16 output)
  launch_gemm<false, false, true, true>(memory, vp_w, vp_b, nullptr, pmask,
                                        valh, MV, DMODEL, DMODEL, stream);
  // 6. sampling offsets
  launch_gemm<false, false, false, false>(tgt1, so_w, so_b, nullptr, nullptr,
                                          offb, BQ, 256, DMODEL, stream);
  // 7. attention weight logits
  launch_gemm<false, false, false, false>(tgt1, aw_w, aw_b, nullptr, nullptr,
                                          awlb, BQ, 128, DMODEL, stream);
  // 8. deformable sampling
  deform_kernel<<<dim3(BQ), 256, 0, stream>>>(refp, valh, offb, awlb, ca);
  // 9. output proj + residual(tgt1)
  launch_gemm<false, true, false, false>(ca, op_w, op_b, tgt1, nullptr, tmp,
                                         BQ, DMODEL, DMODEL, stream);
  // 10. LN2 -> tgt2
  layernorm_kernel<<<lnGrid, 256, 0, stream>>>(tmp, n2g, n2b, tgt2, BQ);
  // 11. FFN up + relu
  launch_gemm<true, false, false, false>(tgt2, l1_w, l1_b, nullptr, nullptr,
                                         hid, BQ, DFFN, DMODEL, stream);
  // 12. FFN down + residual(tgt2)
  launch_gemm<false, true, false, false>(hid, l2_w, l2_b, tgt2, nullptr, tmp,
                                         BQ, DMODEL, DFFN, stream);
  // 13. LN3 -> output
  layernorm_kernel<<<lnGrid, 256, 0, stream>>>(tmp, n3g, n3b, outp, BQ);

  (void)in_sizes; (void)n_in; (void)out_size; (void)ws_size;
}
